// GNN_GRU_83519934038653
// MI455X (gfx1250) — matrix operand order repaired and checked by reading, not compiled, not run on hardware
//
#include <hip/hip_runtime.h>

typedef __attribute__((ext_vector_type(16))) _Float16 v16h;
typedef __attribute__((ext_vector_type(8)))  _Float16 v8h;
typedef __attribute__((ext_vector_type(8)))  float    v8f;

#define DEV __device__ __forceinline__

// D = A(16x32 f16) * B(32x16 f16) + C(16x16 f32)
#define WMMA_F16(A, B, C) \
  __builtin_amdgcn_wmma_f32_16x16x32_f16(false, (A), false, (B), (short)0, (C), false, false)

namespace {
constexpr int kN = 10000;            // nodes
constexpr int kE = 160000;           // edges
constexpr int kD = 32;               // hidden dim
constexpr int kSteps = 3;
constexpr int kEdgeTiles = kE / 16;  // 10000 (exact)
constexpr int kNodeTiles = (kN + 15) / 16;  // 625 (exact: 625*16 == 10000)
constexpr int kEdgeWaves = 4;        // waves per block in edge-build kernel
}

// ---------------- WMMA fragment loaders (ISA 7.12.2 / 7.12.5 layouts) -------
// A matrix (16xK, f16, one 16x32 K-chunk). Lane L holds row m = L&15.
// half j: j<8 -> K = kb+j ; j>=8 -> K = kb+8+j ; kb = (L>>4)*8.
DEV v16h afrag_from_f32row(const float* __restrict__ row, int lane) {
  int kb = (lane >> 4) * 8;
  v16h f;
#pragma unroll
  for (int j = 0; j < 8; ++j) {
    f[j]     = (_Float16)row[kb + j];
    f[8 + j] = (_Float16)row[kb + 16 + j];
  }
  return f;
}
DEV v16h afrag_from_h16row(const _Float16* __restrict__ row, int koff, int lane) {
  int kb = koff + (lane >> 4) * 8;
  v16h f;
#pragma unroll
  for (int j = 0; j < 8; ++j) {
    f[j]     = row[kb + j];
    f[8 + j] = row[kb + 16 + j];
  }
  return f;
}
// B matrix (32x16 f16 chunk) from row-major f32 W[K][ncols].
// Lane L holds col n = ntile*16 + (L&15); half j: j<8 -> K = kb+j ; j>=8 -> K = kb+8+j ; kb = (L>>4)*8.
DEV v16h bfrag_from_f32(const float* __restrict__ W, int ncols, int ntile, int lane) {
  int n  = ntile * 16 + (lane & 15);
  int kb = (lane >> 4) * 8;
  v16h f;
#pragma unroll
  for (int j = 0; j < 16; ++j) f[j] = (_Float16)W[(kb + j + (j >= 8 ? 8 : 0)) * ncols + n];
  return f;
}

DEV float sigmoid_f(float x) { return 1.f / (1.f + __expf(-x)); }
DEV float tanh_f(float x)    { float e = __expf(-2.f * x); return (1.f - e) / (1.f + e); }

// ---------------- kernel 0: node embedding h = x @ W_emb + b_emb ------------
__global__ void embed_kernel(const float* __restrict__ x,
                             const float* __restrict__ W_emb,
                             const float* __restrict__ b_emb,
                             float* __restrict__ h) {
  int i = blockIdx.x * blockDim.x + threadIdx.x;
  if (i >= kN * kD) return;
  int n = i >> 5, d = i & 31;
  h[i] = x[n * 2] * W_emb[d] + x[n * 2 + 1] * W_emb[kD + d] + b_emb[d];
}

__global__ void zero_kernel(float* __restrict__ p, int n) {
  int i = blockIdx.x * blockDim.x + threadIdx.x;
  if (i < n) p[i] = 0.f;
}

// ---------------- kernel 1: pack W2 (f32 64x1024) into f16 B-fragment order -
// Layout: frag (nt in 0..63, kchunk c in 0..1) -> 32 lanes x 16 halves contiguous,
// so a wave loads each B operand as two global_load_b128 per lane.
__global__ void w2_pack_kernel(const float* __restrict__ W2,
                               _Float16* __restrict__ W2frag) {
  int idx = blockIdx.x * blockDim.x + threadIdx.x;  // 0 .. 65535
  if (idx >= 64 * 2 * 32 * 16) return;
  int j    = idx & 15;
  int lane = (idx >> 4) & 31;
  int c    = (idx >> 9) & 1;
  int nt   = idx >> 10;
  int k = c * 32 + ((lane >> 4) * 8) + j + (j >= 8 ? 8 : 0);
  int n = nt * 16 + (lane & 15);
  W2frag[idx] = (_Float16)W2[k * 1024 + n];
}

// ---------------- kernel 2: fused per-edge pipeline -> We (f16) -------------
// One wave handles 16 edges: he (2x2 WMMA) -> LDS -> h1 (4 WMMA) -> LDS ->
// We = h1@W2+b2 (64 tiles x 2 WMMA), stored f16 row-major [e][i*32+o].
__global__ void edge_build_kernel(const float* __restrict__ h,
                                  const int* __restrict__ src,
                                  const int* __restrict__ dst,
                                  const float* __restrict__ W_theta,
                                  const float* __restrict__ b_theta,
                                  const float* __restrict__ W_phi,
                                  const float* __restrict__ b_phi,
                                  const float* __restrict__ W1,
                                  const float* __restrict__ b1,
                                  const _Float16* __restrict__ W2frag,
                                  const float* __restrict__ b2,
                                  _Float16* __restrict__ We) {
  __shared__ _Float16 lds_he[kEdgeWaves][16 * 32];
  __shared__ _Float16 lds_h1[kEdgeWaves][16 * 64];

  int lane = threadIdx.x & 31;
  int w    = threadIdx.x >> 5;
  int tile = blockIdx.x * kEdgeWaves + w;   // grid covers kEdgeTiles exactly
  int m    = lane & 15;
  int hi16 = (lane >> 4) << 3;              // 0 or 8 (C-layout row offset)

  // gather this lane's edge endpoints (A rows = edges)
  int e_m = tile * 16 + m;
  int s = src[e_m], d = dst[e_m];
  const float* hs = h + (size_t)s * kD;
  const float* hd = h + (size_t)d * kD;
  v16h a_hs, a_df;
  {
    int kb = (lane >> 4) * 8;
#pragma unroll
    for (int j = 0; j < 8; ++j) {
      float vs0 = hs[kb + j],      vd0 = hd[kb + j];
      float vs1 = hs[kb + 16 + j], vd1 = hd[kb + 16 + j];
      a_hs[j]     = (_Float16)vs0;  a_df[j]     = (_Float16)(vd0 - vs0);
      a_hs[8 + j] = (_Float16)vs1;  a_df[8 + j] = (_Float16)(vd1 - vs1);
    }
  }

  // he = relu((hd-hs)@W_theta + hs@W_phi + b_theta + b_phi)   [16 x 32]
#pragma unroll
  for (int nt = 0; nt < 2; ++nt) {
    v16h Bt = bfrag_from_f32(W_theta, kD, nt, lane);
    v16h Bp = bfrag_from_f32(W_phi,   kD, nt, lane);
    v8f c = {};
    c = WMMA_F16(a_df, Bt, c);
    c = WMMA_F16(a_hs, Bp, c);
    int n = nt * 16 + (lane & 15);
    float bias = b_theta[n] + b_phi[n];
#pragma unroll
    for (int r = 0; r < 8; ++r) {
      float v = c[r] + bias;
      v = v > 0.f ? v : 0.f;
      lds_he[w][(r + hi16) * 32 + n] = (_Float16)v;
    }
  }
  __syncthreads();

  v16h a_he = afrag_from_h16row(&lds_he[w][m * 32], 0, lane);

  // h1 = relu(he @ W1 + b1)   [16 x 64]
#pragma unroll
  for (int nt = 0; nt < 4; ++nt) {
    v16h B = bfrag_from_f32(W1, 64, nt, lane);
    v8f c = {};
    c = WMMA_F16(a_he, B, c);
    int n = nt * 16 + (lane & 15);
    float bias = b1[n];
#pragma unroll
    for (int r = 0; r < 8; ++r) {
      float v = c[r] + bias;
      v = v > 0.f ? v : 0.f;
      lds_h1[w][(r + hi16) * 64 + n] = (_Float16)v;
    }
  }
  __syncthreads();

  v16h a0 = afrag_from_h16row(&lds_h1[w][m * 64], 0,  lane);
  v16h a1 = afrag_from_h16row(&lds_h1[w][m * 64], 32, lane);

  // We = h1 @ W2 + b2   [16 x 1024], stored f16, non-temporal (327MB > L2)
  _Float16* Wout = We + (size_t)tile * 16 * 1024;
  for (int nt = 0; nt < 64; ++nt) {
    v16h B0 = *(const v16h*)(W2frag + ((size_t)(nt * 2 + 0) * 32 + lane) * 16);
    v16h B1 = *(const v16h*)(W2frag + ((size_t)(nt * 2 + 1) * 32 + lane) * 16);
    v8f c = {};
    c = WMMA_F16(a0, B0, c);
    c = WMMA_F16(a1, B1, c);
    int n = nt * 16 + (lane & 15);
    float bias = b2[n];
#pragma unroll
    for (int r = 0; r < 8; ++r) {
      __builtin_nontemporal_store((_Float16)(c[r] + bias),
                                  &Wout[(size_t)(r + hi16) * 1024 + n]);
    }
  }
}

// ---------------- kernel 3: msg = h[src] x We[e]; scatter-add into a --------
// One wave per edge. Each lane streams 8 f16 of We per iter (wave reads 512
// contiguous bytes per global_load_b128 group), NT hint to bypass L2 retention.
__global__ void msg_kernel(const float* __restrict__ h,
                           const int* __restrict__ src,
                           const int* __restrict__ dst,
                           const _Float16* __restrict__ We,
                           float* __restrict__ a_buf) {
  int lane = threadIdx.x & 31;
  int e = blockIdx.x * (blockDim.x >> 5) + (threadIdx.x >> 5);
  int s = src[e], d = dst[e];
  float hv = h[(size_t)s * kD + lane];          // lane holds h[src][lane]
  const _Float16* Wrow = We + (size_t)e * 1024;

  int row0 = lane >> 2;        // i offset within each 8-row block
  int ob   = (lane & 3) * 8;   // output-col base handled by this lane
  float acc[8];
#pragma unroll
  for (int j = 0; j < 8; ++j) acc[j] = 0.f;

#pragma unroll
  for (int ii = 0; ii < 4; ++ii) {
    int i = ii * 8 + row0;
    float a = __shfl(hv, i, 32);               // broadcast h[src][i]
    v8h vals = __builtin_nontemporal_load(
        (const v8h*)(Wrow + (size_t)ii * 256 + (size_t)lane * 8));
#pragma unroll
    for (int j = 0; j < 8; ++j) acc[j] += a * (float)vals[j];
  }
  // reduce over lanes sharing the same (lane&3): xor over bits 2,3,4
#pragma unroll
  for (int j = 0; j < 8; ++j) {
    acc[j] += __shfl_xor(acc[j], 4, 32);
    acc[j] += __shfl_xor(acc[j], 8, 32);
    acc[j] += __shfl_xor(acc[j], 16, 32);
  }
  if (lane < 4) {
    float* out = a_buf + (size_t)d * kD + ob;
#pragma unroll
    for (int j = 0; j < 8; ++j) atomicAdd(&out[j], acc[j]);
  }
}

// ---------------- kernel 4: GRU cell via WMMA (16 nodes per wave) -----------
__global__ void gru_kernel(const float* __restrict__ a_buf,
                           const float* __restrict__ h_in,
                           const float* __restrict__ W_ih,
                           const float* __restrict__ b_ih,
                           const float* __restrict__ W_hh,
                           const float* __restrict__ b_hh,
                           float* __restrict__ h_out) {
  int lane = threadIdx.x & 31;
  int tile = blockIdx.x * (blockDim.x >> 5) + (threadIdx.x >> 5);
  int mb = tile * 16;
  if (mb >= kN) return;                         // wave-uniform guard
  int m    = lane & 15;
  int hi16 = (lane >> 4) << 3;

  v16h Aa = afrag_from_f32row(a_buf + (size_t)(mb + m) * kD, lane);
  v16h Ah = afrag_from_f32row(h_in  + (size_t)(mb + m) * kD, lane);

  v8f gi[6], gh[6];
#pragma unroll
  for (int nt = 0; nt < 6; ++nt) {
    {
      v16h B = bfrag_from_f32(W_ih, 96, nt, lane);
      v8f c = {};
      gi[nt] = WMMA_F16(Aa, B, c);
      float b = b_ih[nt * 16 + (lane & 15)];
#pragma unroll
      for (int r = 0; r < 8; ++r) gi[nt][r] = gi[nt][r] + b;
    }
    {
      v16h B = bfrag_from_f32(W_hh, 96, nt, lane);
      v8f c = {};
      gh[nt] = WMMA_F16(Ah, B, c);
      float b = b_hh[nt * 16 + (lane & 15)];
#pragma unroll
      for (int r = 0; r < 8; ++r) gh[nt][r] = gh[nt][r] + b;
    }
  }

#pragma unroll
  for (int t = 0; t < 2; ++t) {                 // two 16-col halves of each gate
    int col = t * 16 + (lane & 15);
#pragma unroll
    for (int r = 0; r < 8; ++r) {
      float rg = sigmoid_f(gi[t][r]     + gh[t][r]);
      float zg = sigmoid_f(gi[2 + t][r] + gh[2 + t][r]);
      float ng = tanh_f(gi[4 + t][r] + rg * gh[4 + t][r]);
      size_t idx = (size_t)(mb + r + hi16) * kD + col;
      float ho = h_in[idx];
      h_out[idx] = (1.f - zg) * ng + zg * ho;
    }
  }
}

// ---------------- launch ----------------------------------------------------
extern "C" void kernel_launch(void* const* d_in, const int* in_sizes, int n_in,
                              void* d_out, int out_size, void* d_ws, size_t ws_size,
                              hipStream_t stream) {
  (void)in_sizes; (void)n_in; (void)out_size; (void)ws_size;
  const float* x       = (const float*)d_in[0];
  const int*   src     = (const int*)d_in[1];
  const int*   dst     = (const int*)d_in[2];
  const float* W_emb   = (const float*)d_in[3];
  const float* b_emb   = (const float*)d_in[4];
  const float* W_theta = (const float*)d_in[5];
  const float* b_theta = (const float*)d_in[6];
  const float* W_phi   = (const float*)d_in[7];
  const float* b_phi   = (const float*)d_in[8];
  const float* W1      = (const float*)d_in[9];
  const float* b1      = (const float*)d_in[10];
  const float* W2      = (const float*)d_in[11];
  const float* b2      = (const float*)d_in[12];
  const float* W_ih    = (const float*)d_in[13];
  const float* b_ih    = (const float*)d_in[14];
  const float* W_hh    = (const float*)d_in[15];
  const float* b_hh    = (const float*)d_in[16];

  char* ws = (char*)d_ws;
  size_t off = 0;
  float* hA = (float*)(ws + off);        off += (size_t)kN * kD * sizeof(float);
  float* hB = (float*)(ws + off);        off += (size_t)kN * kD * sizeof(float);
  float* a_buf = (float*)(ws + off);     off += (size_t)kN * kD * sizeof(float);
  _Float16* W2frag = (_Float16*)(ws + off); off += (size_t)65536 * sizeof(_Float16);
  off = (off + 255) & ~(size_t)255;
  _Float16* We = (_Float16*)(ws + off);  // kE * 1024 f16 = 327.68 MB

  embed_kernel<<<(kN * kD + 255) / 256, 256, 0, stream>>>(x, W_emb, b_emb, hA);
  w2_pack_kernel<<<65536 / 256, 256, 0, stream>>>(W2, W2frag);
  edge_build_kernel<<<kEdgeTiles / kEdgeWaves, kEdgeWaves * 32, 0, stream>>>(
      hA, src, dst, W_theta, b_theta, W_phi, b_phi, W1, b1, W2frag, b2, We);

  float* hcur = hA;
  float* halt = hB;
  for (int step = 0; step < kSteps; ++step) {
    zero_kernel<<<(kN * kD + 255) / 256, 256, 0, stream>>>(a_buf, kN * kD);
    msg_kernel<<<kE / 8, 256, 0, stream>>>(hcur, src, dst, We, a_buf);
    float* hout = (step == kSteps - 1) ? (float*)d_out : halt;
    gru_kernel<<<(kNodeTiles + 3) / 4, 128, 0, stream>>>(
        a_buf, hcur, W_ih, b_ih, W_hh, b_hh, hout);
    float* tmp = hcur; hcur = halt; halt = tmp;
    hcur = (step == kSteps - 1) ? hcur : hcur;  // (swap; last iter unused)
    if (step < kSteps - 1) { /* hcur now points at the freshly written buffer */ }
  }
}
